// GNNEncoder_46703474376723
// MI455X (gfx1250) — matrix operand order repaired and checked by reading, not compiled, not run on hardware
//
#include <hip/hip_runtime.h>
#include <hip/hip_bf16.h>

#define IN_CH  64
#define HID_CH 128

typedef __attribute__((ext_vector_type(2)))  float  v2f;
typedef __attribute__((ext_vector_type(8)))  float  v8f;
typedef __attribute__((ext_vector_type(16))) __bf16 v16bf;

// ---------------------------------------------------------------------------
// scratch helpers
// ---------------------------------------------------------------------------
__global__ void zero_kernel(float* __restrict__ p, long long n) {
    long long i = (long long)blockIdx.x * blockDim.x + threadIdx.x;
    if (i < n) p[i] = 0.0f;
}

__global__ void degree_kernel(const int* __restrict__ dst,
                              float* __restrict__ cnt, int nE) {
    int e = blockIdx.x * blockDim.x + threadIdx.x;
    if (e < nE) atomicAdd(&cnt[dst[e]], 1.0f);
}

// one thread per (edge, 4-channel group): float4 gather + 4 global f32 atomics
template <int C>
__global__ void scatter_kernel(const int* __restrict__ src,
                               const int* __restrict__ dst,
                               const float* __restrict__ feat,
                               float* __restrict__ agg, int nE) {
    const int G = C / 4;
    long long gid = (long long)blockIdx.x * blockDim.x + threadIdx.x;
    if (gid >= (long long)nE * G) return;
    int e = (int)(gid / G);
    int c = (int)(gid % G) * 4;
    long long s = src[e], d = dst[e];
    float4 v = *(const float4*)(feat + s * C + c);
    float* p = agg + d * C + c;
    atomicAdd(p + 0, v.x);
    atomicAdd(p + 1, v.y);
    atomicAdd(p + 2, v.z);
    atomicAdd(p + 3, v.w);
}

template <int C>
__global__ void mean_kernel(float* __restrict__ agg,
                            const float* __restrict__ cnt, long long n) {
    long long i = (long long)blockIdx.x * blockDim.x + threadIdx.x;
    if (i < n) agg[i] = agg[i] / fmaxf(cnt[i / C], 1.0f);
}

// ---------------------------------------------------------------------------
// SAGE GEMM: out[N,128] = Am[N,K] @ Wl[K,128] + Xs[N,K] @ Wr[K,128] + bias
// one wave -> one 16x16 tile of out; 8 waves/block span all 128 columns.
// Requires N % 16 == 0 (100000 = 6250*16) so EXEC is all-ones for WMMA.
// ---------------------------------------------------------------------------
#define USE_F32_WMMA __has_builtin(__builtin_amdgcn_wmma_f32_16x16x4_f32)

template <int K, bool RELU>
__launch_bounds__(256)
__global__ void sage_gemm_kernel(const float* __restrict__ Am,
                                 const float* __restrict__ Xs,
                                 const float* __restrict__ Wl,
                                 const float* __restrict__ Wr,
                                 const float* __restrict__ bias,
                                 float* __restrict__ out) {
    const int lane = threadIdx.x & 31;
    const int wave = threadIdx.x >> 5;      // 0..7 -> N-tile
    const int hi   = lane >> 4;             // 0 or 1 (lane half)
    const int col  = lane & 15;
    const int row  = blockIdx.x * 16 + col; // A-matrix row for this lane
    const int n0   = wave * 16;

    v8f acc = {};

#if USE_F32_WMMA
    // FP32 WMMA 16x16x4.  A 16x4: v0 = K(0|2), v1 = K(1|3) split on lane half.
    // B 4x16 mirrors:      v0 = row K(0|2), v1 = row K(1|3), N = lane&15.
    const int kh = hi * 2;
    for (int k = 0; k < K; k += 4) {
        v2f a = *(const v2f*)(Am + (long long)row * K + k + kh);
        v2f b;
        b.x = Wl[(k + kh) * HID_CH + n0 + col];
        b.y = Wl[(k + kh + 1) * HID_CH + n0 + col];
        acc = __builtin_amdgcn_wmma_f32_16x16x4_f32(false, a, false, b,
                                                    (short)0, acc, false, false);
        a = *(const v2f*)(Xs + (long long)row * K + k + kh);
        b.x = Wr[(k + kh) * HID_CH + n0 + col];
        b.y = Wr[(k + kh + 1) * HID_CH + n0 + col];
        acc = __builtin_amdgcn_wmma_f32_16x16x4_f32(false, a, false, b,
                                                    (short)0, acc, false, false);
    }
#else
    // Fallback: BF16 WMMA 16x16x32 with fp32 accumulate (codegen-confirmed).
    // A 16x32 layout: elem i -> K = (i&7) + 16*(i>>3) + 8*hi
    // B 32x16 layout: elem i -> K = (i&7) + 16*(i>>3) + 8*hi, N = lane&15
    for (int k = 0; k < K; k += 32) {
        v16bf a, b;
        for (int i = 0; i < 16; ++i) {
            int kk = k + (i & 7) + 16 * (i >> 3) + 8 * hi;
            a[i] = (__bf16)Am[(long long)row * K + kk];
        }
        for (int i = 0; i < 16; ++i) {
            int kk = k + (i & 7) + 16 * (i >> 3) + 8 * hi;
            b[i] = (__bf16)Wl[kk * HID_CH + n0 + col];
        }
        acc = __builtin_amdgcn_wmma_f32_16x16x32_bf16(false, a, false, b,
                                                      (short)0, acc, false, false);
        for (int i = 0; i < 16; ++i) {
            int kk = k + (i & 7) + 16 * (i >> 3) + 8 * hi;
            a[i] = (__bf16)Xs[(long long)row * K + kk];
        }
        for (int i = 0; i < 16; ++i) {
            int kk = k + (i & 7) + 16 * (i >> 3) + 8 * hi;
            b[i] = (__bf16)Wr[kk * HID_CH + n0 + col];
        }
        acc = __builtin_amdgcn_wmma_f32_16x16x32_bf16(false, a, false, b,
                                                      (short)0, acc, false, false);
    }
#endif

    // C/D layout: VGPR r, lane half hi -> M = r + 8*hi, N = lane&15
    const float bv = bias[n0 + col];
    const int mbase = blockIdx.x * 16 + 8 * hi;
    for (int r = 0; r < 8; ++r) {
        float v = acc[r] + bv;
        if (RELU) v = fmaxf(v, 0.0f);
        out[(long long)(mbase + r) * HID_CH + n0 + col] = v;
    }
}

// ---------------------------------------------------------------------------
extern "C" void kernel_launch(void* const* d_in, const int* in_sizes, int n_in,
                              void* d_out, int out_size, void* d_ws, size_t ws_size,
                              hipStream_t stream) {
    const float* x   = (const float*)d_in[0];
    const int*   ei  = (const int*)d_in[1];   // edge_index [2, E] (harness ABI: int)
    const float* W1l = (const float*)d_in[2];
    const float* b1  = (const float*)d_in[3];
    const float* W1r = (const float*)d_in[4];
    const float* W2l = (const float*)d_in[5];
    const float* b2  = (const float*)d_in[6];
    const float* W2r = (const float*)d_in[7];
    float* out = (float*)d_out;

    const int nN = in_sizes[0] / IN_CH;       // 100000
    const int nE = in_sizes[1] / 2;           // 1250000
    const int* src = ei;
    const int* dst = ei + nE;

    // workspace: cnt[N] | agg[N,128] | h[N,128]   (~103 MB)
    char* ws = (char*)d_ws;
    size_t off = 0;
    float* cnt = (float*)(ws + off);
    off += (((size_t)nN * 4) + 255) & ~(size_t)255;
    float* agg = (float*)(ws + off);
    off += (((size_t)nN * HID_CH * 4) + 255) & ~(size_t)255;
    float* h = (float*)(ws + off);

    auto cdiv = [](long long a, long long b) { return (int)((a + b - 1) / b); };

    // degrees (shared by both layers)
    zero_kernel<<<cdiv(nN, 256), 256, 0, stream>>>(cnt, nN);
    degree_kernel<<<cdiv(nE, 256), 256, 0, stream>>>(dst, cnt, nE);

    // ---- layer 1: h = relu(mean(x) @ W1l + b1 + x @ W1r) ----
    zero_kernel<<<cdiv((long long)nN * IN_CH, 256), 256, 0, stream>>>(
        agg, (long long)nN * IN_CH);
    scatter_kernel<IN_CH><<<cdiv((long long)nE * (IN_CH / 4), 256), 256, 0, stream>>>(
        src, dst, x, agg, nE);
    mean_kernel<IN_CH><<<cdiv((long long)nN * IN_CH, 256), 256, 0, stream>>>(
        agg, cnt, (long long)nN * IN_CH);
    sage_gemm_kernel<IN_CH, true><<<nN / 16, 256, 0, stream>>>(
        agg, x, W1l, W1r, b1, h);

    // ---- layer 2: out = mean(h) @ W2l + b2 + h @ W2r ----
    zero_kernel<<<cdiv((long long)nN * HID_CH, 256), 256, 0, stream>>>(
        agg, (long long)nN * HID_CH);
    scatter_kernel<HID_CH><<<cdiv((long long)nE * (HID_CH / 4), 256), 256, 0, stream>>>(
        src, dst, h, agg, nE);
    mean_kernel<HID_CH><<<cdiv((long long)nN * HID_CH, 256), 256, 0, stream>>>(
        agg, cnt, (long long)nN * HID_CH);
    sage_gemm_kernel<HID_CH, false><<<nN / 16, 256, 0, stream>>>(
        agg, h, W2l, W2r, b2, out);
}
